// testResNet_76115410419909
// MI455X (gfx1250) — matrix operand order repaired and checked by reading, not compiled, not run on hardware
//
#include <hip/hip_runtime.h>
#include <hip/hip_bf16.h>

// ---------------------------------------------------------------------------
// ResNet-20 forward for MI455X (gfx1250, wave32).
// Convs: implicit GEMM on v_wmma_f32_16x16x32_f16, weights staged in LDS.
// Activations f16 NHWC; BN statistics are computed in-register from the WMMA
// accumulators (no extra full-tensor pass) -> minimal HBM traffic @ 23.3 TB/s.
// ---------------------------------------------------------------------------

typedef __attribute__((ext_vector_type(16))) _Float16 v16h;
typedef __attribute__((ext_vector_type(8)))  float    v8f;

union Frag16 { v16h h; unsigned u[8]; };

// --------------------------- weight generation -----------------------------
__global__ __launch_bounds__(256) void wflat_kernel(
    const float* __restrict__ ow, const float* __restrict__ P,
    const float* __restrict__ np, float* __restrict__ wflat, int D)
{
    int d = blockIdx.x * 256 + threadIdx.x;
    if (d >= D) return;
    float s = ow[d];
#pragma unroll 8
    for (int l = 0; l < 40; ++l)
        s = fmaf(np[l], P[(size_t)l * D + d], s);
    wflat[d] = s;
}

// Pack OIHW f32 weights -> [Cout][Kpad] f16, K = (kh*3+kw)*Cin + cin, zero pad.
__global__ __launch_bounds__(256) void pack_w_kernel(
    const float* __restrict__ wflat, _Float16* __restrict__ dst,
    int srcOff, int Cin, int Cout, int Kpad)
{
    int i = blockIdx.x * 256 + threadIdx.x;
    if (i >= Cout * Kpad) return;
    int o = i / Kpad, k = i % Kpad;
    float v = 0.f;
    if (k < Cin * 9) {
        int kpos = k / Cin, cin = k % Cin;
        int kh = kpos / 3, kw = kpos % 3;
        v = wflat[srcOff + (((o * Cin + cin) * 3 + kh) * 3 + kw)];
    }
    dst[i] = (_Float16)v;
}

// NCHW f32 input -> NHWC f16
__global__ __launch_bounds__(256) void cvt_in_kernel(
    const float* __restrict__ x, _Float16* __restrict__ dst)
{
    long i = (long)blockIdx.x * 256 + threadIdx.x;
    const long total = (long)1024 * 32 * 32 * 3;
    if (i >= total) return;
    int c = (int)(i % 3);
    long m = i / 3;
    int w = (int)(m % 32);
    int h = (int)((m / 32) % 32);
    int n = (int)(m / 1024);
    dst[i] = (_Float16)x[(((size_t)n * 3 + c) * 32 + h) * 32 + w];
}

// ------------------------------ conv (WMMA) --------------------------------
// Implicit GEMM, all shape parameters compile-time.
//   M = 1024*WOUT*WOUT pixels, N = COUT = NT*16, K = CIN*9 padded to 32.
// 8 waves/block, each wave owns one 16-pixel M-tile and all NT cout tiles.
// Weights live in LDS; A fragments are two b128 global gathers per k-chunk.
// Output stored f16; per-block per-channel (sum, sumsq) partials emitted for
// BN (computed from f32 accumulators, fixed reduction order -> deterministic).
template<int CIN, int STRIDE, int WOUT, int NT>
__global__ __launch_bounds__(256) void conv_wmma_kernel(
    const _Float16* __restrict__ act, const _Float16* __restrict__ wgt,
    _Float16* __restrict__ outp, float* __restrict__ partial)
{
    constexpr int KREAL = CIN * 9;
    constexpr int KP    = (KREAL + 31) & ~31;
    constexpr int COUT  = NT * 16;
    constexpr int HIN   = WOUT * STRIDE;
    constexpr int LW    = (WOUT == 32) ? 5 : (WOUT == 16) ? 4 : 3;

    __shared__ _Float16 wl[COUT * KP];
    __shared__ float    red[256 * NT * 2];

    // cooperative weight stage: global -> LDS, 16B vectors
    {
        const uint4* src = (const uint4*)wgt;
        uint4*       dst = (uint4*)wl;
        constexpr int NV = COUT * KP / 8;
        for (int i = threadIdx.x; i < NV; i += 256) dst[i] = src[i];
    }
    __syncthreads();

    const int tid  = threadIdx.x;
    const int lane = tid & 31;
    const int wave = tid >> 5;
    const int hi   = lane >> 4;                    // 0 or 1
    const long tile_m = ((long)blockIdx.x * 8 + wave) * 16;

    // pixel owned by this lane for the A fragment
    const long m  = tile_m + (lane & 15);
    const int ow  = (int)(m & (WOUT - 1));
    const int oh  = (int)((m >> LW) & (WOUT - 1));
    const int n   = (int)(m >> (2 * LW));
    const int hbase = oh * STRIDE - 1;
    const int wbase = ow * STRIDE - 1;

    v8f acc[NT];
#pragma unroll
    for (int t = 0; t < NT; ++t) acc[t] = {};

    for (int kc = 0; kc < KP; kc += 32) {
        Frag16 A;
        if constexpr (CIN == 3) {
            // stem layer: scalar gather (single k-chunk, K=27)
#pragma unroll
            for (int v = 0; v < 8; ++v) {
                int ka = kc + ((v < 4) ? 2 * v : 8 + 2 * v) + hi * 8;
                unsigned short h0 = 0, h1 = 0;
#pragma unroll
                for (int e = 0; e < 2; ++e) {
                    int kk = ka + e;
                    if (kk < KREAL) {
                        int kpos = kk / CIN, cin = kk - kpos * CIN;
                        int ih = hbase + kpos / 3, iw = wbase + kpos % 3;
                        if (ih >= 0 && ih < HIN && iw >= 0 && iw < HIN) {
                            union { _Float16 f; unsigned short u; } cv;
                            cv.f = act[(((size_t)n * HIN + ih) * HIN + iw) * CIN + cin];
                            if (e == 0) h0 = cv.u; else h1 = cv.u;
                        }
                    }
                }
                A.u[v] = (unsigned)h0 | ((unsigned)h1 << 16);
            }
        } else {
            // Two 16B groups: VGPRs v0..3 hold K0=kc+hi*8+[0,8),
            // v4..7 hold K0+16+[0,8). 8|CIN and 8|KREAL: each group is
            // 8 contiguous f16 in NHWC memory under one (kh,kw) tap.
#pragma unroll
            for (int g = 0; g < 2; ++g) {
                const int K0 = kc + g * 16 + hi * 8;
                uint4 val = make_uint4(0u, 0u, 0u, 0u);
                if (K0 < KREAL) {
                    const int kpos = K0 / CIN, cin = K0 - kpos * CIN;
                    const int ih = hbase + kpos / 3, iw = wbase + kpos % 3;
                    if (ih >= 0 && ih < HIN && iw >= 0 && iw < HIN)
                        val = *(const uint4*)(act +
                              (((size_t)n * HIN + ih) * HIN + iw) * CIN + cin);
                }
                A.u[4 * g + 0] = val.x; A.u[4 * g + 1] = val.y;
                A.u[4 * g + 2] = val.z; A.u[4 * g + 3] = val.w;
            }
        }

#pragma unroll
        for (int t = 0; t < NT; ++t) {
            // B fragment: [kc+hi*8, +8) and [kc+16+hi*8, +8) of weight row -> 2x ds b128
            const _Float16* bp =
                &wl[(size_t)(t * 16 + (lane & 15)) * KP + kc + hi * 8];
            uint4 b0 = *(const uint4*)bp;
            uint4 b1 = *(const uint4*)(bp + 16);
            Frag16 B;
            B.u[0] = b0.x; B.u[1] = b0.y; B.u[2] = b0.z; B.u[3] = b0.w;
            B.u[4] = b1.x; B.u[5] = b1.y; B.u[6] = b1.z; B.u[7] = b1.w;
            acc[t] = __builtin_amdgcn_wmma_f32_16x16x32_f16(
                         false, A.h, false, B.h, (short)0, acc[t], false, false);
        }
    }

    // D layout: VGPR r -> M = r + hi*8, N = lane&15. Store as f16.
    const int mofs = hi * 8;
#pragma unroll
    for (int t = 0; t < NT; ++t) {
        _Float16* orow = outp + (size_t)tile_m * COUT + t * 16 + (lane & 15);
#pragma unroll
        for (int r = 0; r < 8; ++r)
            orow[(size_t)(r + mofs) * COUT] = (_Float16)acc[t][r];
    }

    // ---- per-block BN partials (fixed-order reduction, deterministic) ----
#pragma unroll
    for (int t = 0; t < NT; ++t) {
        float s = 0.f, s2 = 0.f;
#pragma unroll
        for (int r = 0; r < 8; ++r) { float v = acc[t][r]; s += v; s2 += v * v; }
        red[(tid * NT + t) * 2 + 0] = s;
        red[(tid * NT + t) * 2 + 1] = s2;
    }
    __syncthreads();
    if (tid < COUT) {
        const int c = tid, t = c >> 4, cl = c & 15;
        float a = 0.f, b = 0.f;
#pragma unroll
        for (int w8 = 0; w8 < 8; ++w8)
#pragma unroll
            for (int h2 = 0; h2 < 2; ++h2) {
                const int src = w8 * 32 + h2 * 16 + cl;
                a += red[(src * NT + t) * 2 + 0];
                b += red[(src * NT + t) * 2 + 1];
            }
        partial[(size_t)blockIdx.x * COUT * 2 + c * 2 + 0] = a;
        partial[(size_t)blockIdx.x * COUT * 2 + c * 2 + 1] = b;
    }
}

// ------------------- BN stats finalize (tiny, deterministic) ---------------
__global__ __launch_bounds__(256) void stats_finalize_kernel(
    const float* __restrict__ partial, float* __restrict__ st, int NB, int Cout)
{
    __shared__ float s1[256], s2[256];
    const int c = blockIdx.x, t = threadIdx.x;
    float a = 0.f, b = 0.f;
    for (int i = t; i < NB; i += 256) {
        a += partial[(size_t)i * Cout * 2 + c * 2 + 0];
        b += partial[(size_t)i * Cout * 2 + c * 2 + 1];
    }
    s1[t] = a; s2[t] = b;
    __syncthreads();
    for (int o = 128; o > 0; o >>= 1) {
        if (t < o) { s1[t] += s1[t + o]; s2[t] += s2[t + o]; }
        __syncthreads();
    }
    if (t == 0) { st[2 * c] = s1[0]; st[2 * c + 1] = s2[0]; }
}

// ----------------------- BN apply (+ReLU) -> f16 ---------------------------
__global__ __launch_bounds__(256) void bn_relu_kernel(
    const _Float16* __restrict__ conv, const float* __restrict__ st,
    _Float16* __restrict__ dst, long M, int Cout)
{
    long i = (long)blockIdx.x * 256 + threadIdx.x;
    if (i >= M * (long)Cout) return;
    int c = (int)(i % Cout);
    float cnt  = (float)M;
    float mean = st[2 * c] / cnt;
    float var  = st[2 * c + 1] / cnt - mean * mean;
    float y = ((float)conv[i] - mean) * rsqrtf(var + 1e-5f);
    dst[i] = (_Float16)fmaxf(y, 0.f);
}

// ------------- BN apply + residual shortcut + ReLU -> f16 ------------------
__global__ __launch_bounds__(256) void bn_res_relu_kernel(
    const _Float16* __restrict__ conv, const float* __restrict__ st,
    _Float16* __restrict__ dst, const _Float16* __restrict__ prev,
    long M, int Cout, int Hout, int Wout,
    int sc, int padlo, int Cprev, int Hprev, int Wprev)
{
    long i = (long)blockIdx.x * 256 + threadIdx.x;
    if (i >= M * (long)Cout) return;
    int  c = (int)(i % Cout);
    long m = i / Cout;
    float cnt  = (float)M;
    float mean = st[2 * c] / cnt;
    float var  = st[2 * c + 1] / cnt - mean * mean;
    float y = ((float)conv[i] - mean) * rsqrtf(var + 1e-5f);

    int  ow = (int)(m % Wout);
    long on = m / Wout;
    int  oh = (int)(on % Hout);
    int  n  = (int)(on / Hout);
    int  cs = c - padlo;
    if (cs >= 0 && cs < Cprev)
        y += (float)prev[(((size_t)n * Hprev + oh * sc) * Wprev + ow * sc) * Cprev + cs];
    dst[i] = (_Float16)fmaxf(y, 0.f);
}

// ------------------------ avg-pool + FC head -------------------------------
__global__ __launch_bounds__(64) void head_kernel(
    const _Float16* __restrict__ act, const float* __restrict__ Wfc,
    const float* __restrict__ bfc, float* __restrict__ out)
{
    __shared__ float pooled[64];
    const int n = blockIdx.x, t = threadIdx.x;
    float s = 0.f;
#pragma unroll 8
    for (int p = 0; p < 64; ++p)
        s += (float)act[((size_t)n * 64 + p) * 64 + t];
    pooled[t] = s * (1.0f / 64.0f);
    __syncthreads();
    if (t < 10) {
        float a = bfc[t];
#pragma unroll 8
        for (int c = 0; c < 64; ++c)
            a = fmaf(pooled[c], Wfc[t * 64 + c], a);
        out[n * 10 + t] = a;
    }
}

// ---------------------------------------------------------------------------
extern "C" void kernel_launch(void* const* d_in, const int* in_sizes, int n_in,
                              void* d_out, int out_size, void* d_ws, size_t ws_size,
                              hipStream_t stream) {
    const float* x   = (const float*)d_in[0];
    const float* ow  = (const float*)d_in[1];
    const float* P   = (const float*)d_in[2];
    const float* np  = (const float*)d_in[3];
    const float* Wfc = (const float*)d_in[4];
    const float* bfc = (const float*)d_in[5];
    float* out = (float*)d_out;
    (void)in_sizes; (void)n_in; (void)out_size; (void)ws_size;

    const int OC[19] = {16,16,16,16,16,16,16,32,32,32,32,32,32,64,64,64,64,64,64};
    const int IC[19] = { 3,16,16,16,16,16,16,16,32,32,32,32,32,32,64,64,64,64,64};
    int srcOff[19], dstOff[19], Kpad[19];
    int  so = 0; long dsto = 0;
    for (int i = 0; i < 19; ++i) {
        srcOff[i] = so;            so  += OC[i] * IC[i] * 9;
        Kpad[i]   = ((IC[i] * 9 + 31) / 32) * 32;
        dstOff[i] = (int)dsto;     dsto += (long)OC[i] * Kpad[i];
    }
    const int Dtot = so;  // 267696

    auto align256 = [](size_t v) { return (v + 255) & ~(size_t)255; };
    char* p = (char*)d_ws;
    float*    wflat = (float*)p;    p += align256((size_t)Dtot * 4);
    _Float16* wf16  = (_Float16*)p; p += align256((size_t)dsto * 2);
    float*    stats = (float*)p;    p += 256;
    float*    part  = (float*)p;    p += align256((size_t)8192 * 16 * 2 * 4); // 1MB
    _Float16* xin   = (_Float16*)p; p += align256((size_t)1024 * 32 * 32 * 3 * 2);
    const size_t ACT = (size_t)1024 * 32 * 32 * 16;  // max activation elements
    _Float16* bufA  = (_Float16*)p; p += align256(ACT * 2);
    _Float16* bufB  = (_Float16*)p; p += align256(ACT * 2);
    _Float16* bufC  = (_Float16*)p; p += align256(ACT * 2);
    _Float16* convb = (_Float16*)p; p += align256(ACT * 2);

    // 1) reparameterized weights, packed per layer
    wflat_kernel<<<(Dtot + 255) / 256, 256, 0, stream>>>(ow, P, np, wflat, Dtot);
    for (int i = 0; i < 19; ++i) {
        int nel = OC[i] * Kpad[i];
        pack_w_kernel<<<(nel + 255) / 256, 256, 0, stream>>>(
            wflat, wf16 + dstOff[i], srcOff[i], IC[i], OC[i], Kpad[i]);
    }
    // 2) input to NHWC f16
    {
        long nel = (long)1024 * 32 * 32 * 3;
        cvt_in_kernel<<<(unsigned)((nel + 255) / 256), 256, 0, stream>>>(x, xin);
    }

#define LAUNCH_CONV(CIN_, STR_, WOUT_, NT_, li, actp)                          \
    do {                                                                       \
        long M_ = (long)1024 * (WOUT_) * (WOUT_);                              \
        int NB_ = (int)(M_ / 128);                                             \
        conv_wmma_kernel<CIN_, STR_, WOUT_, NT_>                               \
            <<<dim3((unsigned)NB_), 256, 0, stream>>>(                         \
                (actp), wf16 + dstOff[(li)], convb, part);                     \
        stats_finalize_kernel<<<(NT_) * 16, 256, 0, stream>>>(                 \
            part, stats, NB_, (NT_) * 16);                                     \
    } while (0)

    auto bn1 = [&](long M, int c, _Float16* dst) {
        bn_relu_kernel<<<(unsigned)((M * c + 255) / 256), 256, 0, stream>>>(
            convb, stats, dst, M, c);
    };
    auto bn2 = [&](long M, int c, _Float16* dst, const _Float16* prev, int Hout,
                   int sc, int padlo, int Cprev, int Hprev) {
        bn_res_relu_kernel<<<(unsigned)((M * c + 255) / 256), 256, 0, stream>>>(
            convb, stats, dst, prev, M, c, Hout, Hout, sc, padlo, Cprev, Hprev,
            Hprev);
    };

    _Float16* cur = bufA; _Float16* hbuf = bufB; _Float16* nxt = bufC;
    auto rot = [&]() { _Float16* t = cur; cur = nxt; nxt = t; };

    // 3) stem: relu(bn(conv0(x)))
    LAUNCH_CONV(3, 1, 32, 1, 0, xin);
    bn1((long)1024 * 32 * 32, 16, bufA);

    // 4) stage 1: 16ch @ 32x32, blocks of layers (1,2) (3,4) (5,6)
    for (int b = 0; b < 3; ++b) {
        const long M = (long)1024 * 32 * 32;
        LAUNCH_CONV(16, 1, 32, 1, 1 + 2 * b, cur);  bn1(M, 16, hbuf);
        LAUNCH_CONV(16, 1, 32, 1, 2 + 2 * b, hbuf);
        bn2(M, 16, nxt, cur, 32, 1, 0, 16, 32);
        rot();
    }
    // 5) stage 2: 32ch @ 16x16, blocks (7,8) stride2, (9,10), (11,12)
    {
        const long M = (long)1024 * 16 * 16;
        LAUNCH_CONV(16, 2, 16, 2, 7, cur);  bn1(M, 32, hbuf);
        LAUNCH_CONV(32, 1, 16, 2, 8, hbuf);
        bn2(M, 32, nxt, cur, 16, 2, 8, 16, 32);
        rot();
        for (int b = 0; b < 2; ++b) {
            LAUNCH_CONV(32, 1, 16, 2, 9 + 2 * b, cur);  bn1(M, 32, hbuf);
            LAUNCH_CONV(32, 1, 16, 2, 10 + 2 * b, hbuf);
            bn2(M, 32, nxt, cur, 16, 1, 0, 32, 16);
            rot();
        }
    }
    // 6) stage 3: 64ch @ 8x8, blocks (13,14) stride2, (15,16), (17,18)
    {
        const long M = (long)1024 * 8 * 8;
        LAUNCH_CONV(32, 2, 8, 4, 13, cur);  bn1(M, 64, hbuf);
        LAUNCH_CONV(64, 1, 8, 4, 14, hbuf);
        bn2(M, 64, nxt, cur, 8, 2, 16, 32, 16);
        rot();
        for (int b = 0; b < 2; ++b) {
            LAUNCH_CONV(64, 1, 8, 4, 15 + 2 * b, cur);  bn1(M, 64, hbuf);
            LAUNCH_CONV(64, 1, 8, 4, 16 + 2 * b, hbuf);
            bn2(M, 64, nxt, cur, 8, 1, 0, 64, 8);
            rot();
        }
    }
#undef LAUNCH_CONV

    // 7) head
    head_kernel<<<1024, 64, 0, stream>>>(cur, Wfc, bfc, out);
}
